// GEN_31731218382879
// MI455X (gfx1250) — matrix operand order repaired and checked by reading, not compiled, not run on hardware
//
#include <hip/hip_runtime.h>
#include <cstdint>
#include <cstddef>

typedef _Float16 h16;
typedef __attribute__((ext_vector_type(16))) _Float16 v16h;
typedef __attribute__((ext_vector_type(8)))  _Float16 v8h;
typedef __attribute__((ext_vector_type(8)))  float    v8f;

union V16U { v16h v; v8h h[2]; _Float16 e[16]; };

#define DEVINL static __device__ __forceinline__

#define NBATCH 2
#define NPTS   4096
#define MNODES 10000
#define MPAD   10112   /* 79 * 128 */
#define NEDGE  320000
#define DH     128

DEVINL void zero_acc8(v8f* acc) {
#pragma unroll
  for (int i = 0; i < 8; ++i) { v8f z = {}; acc[i] = z; }
}

// Wave-level GEMM: D[16 x 16*NT] += A[16 x 32*KCH] * Wt^T, A row-major f16 in LDS,
// Wt is [out][Kpad] row-major f16 in global (K contiguous per output column).
template<int KCH, int NT>
DEVINL void gemm_lds(const h16* __restrict__ A, int lda,
                     const h16* __restrict__ Wt, int ldw,
                     int rowA, int lane, v8f* acc)
{
  const int kb0 = (lane & 16) ? 8 : 0;    // A: lanes 16-31 hold K+8 / K+24
  const int kbB = (lane & 16) ? 8 : 0;    // B: lanes 16-31 hold K+8 / K+24
  const int col = lane & 15;
#pragma unroll
  for (int kc = 0; kc < KCH; ++kc) {
    V16U a;
    const h16* pa = A + rowA * lda + kc * 32 + kb0;
    a.h[0] = *(const v8h*)(pa);
    a.h[1] = *(const v8h*)(pa + 16);
#pragma unroll
    for (int nt = 0; nt < NT; ++nt) {
      V16U bf;
      const h16* pb = Wt + (size_t)(nt * 16 + col) * ldw + kc * 32 + kbB;
      bf.h[0] = *(const v8h*)(pb);
      bf.h[1] = *(const v8h*)(pb + 16);
      acc[nt] = __builtin_amdgcn_wmma_f32_16x16x32_f16(false, a.v, false, bf.v,
                                                       (short)0, acc[nt], false, false);
    }
  }
}

template<int NT>
DEVINL void epilog_lds(const v8f* acc, const float* __restrict__ bias, bool relu,
                       h16* __restrict__ out, int ldo, int rowBase, int lane)
{
  const int col = lane & 15;
  const int rOff = (lane & 16) ? 8 : 0;
#pragma unroll
  for (int nt = 0; nt < NT; ++nt) {
    float bv = bias[nt * 16 + col];
#pragma unroll
    for (int r = 0; r < 8; ++r) {
      float v = acc[nt][r] + bv;
      if (relu) v = fmaxf(v, 0.f);
      out[(rowBase + rOff + r) * ldo + nt * 16 + col] = (h16)v;
    }
  }
}

// ---------------------------------------------------------------- weight prep
__global__ void k_prep_wt(const float* __restrict__ W, int K, int N,
                          int Kpad, int Npad, h16* __restrict__ out)
{
  int i = blockIdx.x * blockDim.x + threadIdx.x;
  if (i >= Npad * Kpad) return;
  int n = i / Kpad, k = i % Kpad;
  float v = (n < N && k < K) ? W[k * N + n] : 0.f;
  out[i] = (h16)v;
}

// ------------------------------------------------------- per-point softmax stats
// logit(n,m) = 2*p.n_m - |n_m|^2  (per-row -|p|^2 cancels in softmax)
__global__ void __launch_bounds__(256)
k_rowstats(const float* __restrict__ pts, const float* __restrict__ node_pos,
           float* __restrict__ stats)
{
  int pt = blockIdx.x * 8 + (threadIdx.x >> 5);
  int lane = threadIdx.x & 31;
  float px = pts[pt * 2], py = pts[pt * 2 + 1];
  float mx = -3.0e38f, sum = 0.f;
  for (int m = lane; m < MNODES; m += 32) {
    float nx = node_pos[m * 2], ny = node_pos[m * 2 + 1];
    float lg = 2.f * (px * nx + py * ny) - (nx * nx + ny * ny);
    float nm = fmaxf(mx, lg);
    sum = sum * __expf(mx - nm) + __expf(lg - nm);
    mx = nm;
  }
#pragma unroll
  for (int o = 16; o > 0; o >>= 1) {
    float mo = __shfl_xor(mx, o);
    float so = __shfl_xor(sum, o);
    float nm = fmaxf(mx, mo);
    sum = sum * __expf(mx - nm) + so * __expf(mo - nm);
    mx = nm;
  }
  if (lane == 0) { stats[pt * 2] = mx; stats[pt * 2 + 1] = 1.f / sum; }
}

// ------------------------------------------------------------------- encoder
__global__ void __launch_bounds__(256)
k_encoder(const float* __restrict__ sIn, const float* __restrict__ W1,
          const float* __restrict__ b1,
          const h16* __restrict__ Wt2, const float* __restrict__ b2,
          const h16* __restrict__ Wt3, const float* __restrict__ b3,
          h16* __restrict__ embT)
{
  __shared__ __align__(16) h16 A[128 * 128];
  __shared__ __align__(16) h16 Mid[128 * 128];
  const int base = blockIdx.x * 128;
  const int tid = threadIdx.x, lane = tid & 31, wave = tid >> 5;
  for (int i = tid; i < 128 * 128; i += 256) {       // layer 1 (K=2) in VALU
    int r = i >> 7, j = i & 127;
    int row = base + r;
    float v = fmaf(sIn[row * 2], W1[j], fmaf(sIn[row * 2 + 1], W1[128 + j], b1[j]));
    A[r * 128 + j] = (h16)fmaxf(v, 0.f);
  }
  __syncthreads();
  const int rowA = wave * 16 + (lane & 15);
  v8f acc[8]; zero_acc8(acc);
  gemm_lds<4, 8>(A, 128, Wt2, 128, rowA, lane, acc);
  epilog_lds<8>(acc, b2, true, Mid, 128, wave * 16, lane);
  zero_acc8(acc);
  gemm_lds<4, 8>(Mid, 128, Wt3, 128, rowA, lane, acc);
  const int col = lane & 15;
  const int rOff = (lane & 16) ? 8 : 0;
#pragma unroll
  for (int nt = 0; nt < 8; ++nt) {
    float bv = b3[nt * 16 + col];
#pragma unroll
    for (int r = 0; r < 8; ++r) {
      int row = base + wave * 16 + rOff + r;
      int bb = row >> 12, n = row & (NPTS - 1);
      embT[((size_t)bb * DH + nt * 16 + col) * NPTS + n] = (h16)(acc[nt][r] + bv);
    }
  }
}

// ------------------------------------- latents = scores^T @ emb (A built on the fly)
__global__ void __launch_bounds__(256)
k_latents(const float* __restrict__ pts, const float* __restrict__ stats,
          const float* __restrict__ node_pos, const h16* __restrict__ embT,
          float* __restrict__ latents)
{
  __shared__ float4 stg[128];     // (px, py, rowmax, rcpsum) for a chunk of points
  const int b = blockIdx.y, mt = blockIdx.x;
  const int tid = threadIdx.x, lane = tid & 31, wave = tid >> 5;
  const int mRow = mt * 128 + wave * 16 + (lane & 15);
  float nx = 0.f, ny = 0.f, nsq = 0.f, valid = 0.f;
  if (mRow < MNODES) {
    nx = node_pos[mRow * 2]; ny = node_pos[mRow * 2 + 1];
    nsq = nx * nx + ny * ny; valid = 1.f;
  }
  const int kb0 = (lane & 16) ? 8 : 0;
  const int kbB = (lane & 16) ? 8 : 0;
  const int col = lane & 15;
  v8f acc[8]; zero_acc8(acc);
  for (int nb = 0; nb < NPTS / 128; ++nb) {
    __syncthreads();
    if (tid < 128) {
      int gi = b * NPTS + nb * 128 + tid;
      stg[tid] = make_float4(pts[gi * 2], pts[gi * 2 + 1], stats[gi * 2], stats[gi * 2 + 1]);
    }
    __syncthreads();
#pragma unroll
    for (int kc = 0; kc < 4; ++kc) {
      V16U a;
#pragma unroll
      for (int j = 0; j < 8; ++j) {
        float4 p = stg[kc * 32 + kb0 + j];
        a.e[j] = (h16)(__expf(2.f * (p.x * nx + p.y * ny) - nsq - p.z) * p.w * valid);
        float4 p2 = stg[kc * 32 + 16 + kb0 + j];
        a.e[8 + j] = (h16)(__expf(2.f * (p2.x * nx + p2.y * ny) - nsq - p2.z) * p2.w * valid);
      }
#pragma unroll
      for (int nt = 0; nt < 8; ++nt) {
        V16U bf;
        const h16* pb = embT + (size_t)(b * DH + nt * 16 + col) * NPTS + nb * 128 + kc * 32 + kbB;
        bf.h[0] = *(const v8h*)pb;
        bf.h[1] = *(const v8h*)(pb + 16);
        acc[nt] = __builtin_amdgcn_wmma_f32_16x16x32_f16(false, a.v, false, bf.v,
                                                         (short)0, acc[nt], false, false);
      }
    }
  }
  const int rOff = (lane & 16) ? 8 : 0;
#pragma unroll
  for (int nt = 0; nt < 8; ++nt)
#pragma unroll
    for (int r = 0; r < 8; ++r) {
      int m = mt * 128 + wave * 16 + rOff + r;
      if (m < MNODES)
        latents[((size_t)b * MNODES + m) * DH + nt * 16 + col] = acc[nt][r];
    }
}

// --------------------------------------------------- edge message MLP + scatter
__global__ void __launch_bounds__(256)
k_edge(const float* __restrict__ nodes, const int* __restrict__ recvIdx,
       const int* __restrict__ sendIdx,
       const h16* __restrict__ Wt1, const float* __restrict__ b1,
       const h16* __restrict__ Wt2, const float* __restrict__ b2,
       const h16* __restrict__ Wt3, const float* __restrict__ b3,
       float* __restrict__ inbox)
{
  __shared__ __align__(16) h16 A[128 * 128];
  __shared__ __align__(16) h16 Mid[128 * 128];
  const int b = blockIdx.y;
  const int ebase = blockIdx.x * 128;
  const int tid = threadIdx.x, lane = tid & 31, wave = tid >> 5;
  const float* nodesB = nodes + (size_t)b * MNODES * DH;
  float* inb = inbox + (size_t)b * MNODES * DH;

  for (int i = tid; i < 128 * 32; i += 256) {       // stage receiver half (K 0..127)
    int r = i >> 5, c4 = (i & 31) * 4;
    float4 v = *(const float4*)(nodesB + (size_t)recvIdx[ebase + r] * DH + c4);
    h16* dst = A + r * 128 + c4;
    dst[0] = (h16)v.x; dst[1] = (h16)v.y; dst[2] = (h16)v.z; dst[3] = (h16)v.w;
  }
  __syncthreads();
  const int rowA = wave * 16 + (lane & 15);
  v8f acc[8]; zero_acc8(acc);
  gemm_lds<4, 8>(A, 128, Wt1, 256, rowA, lane, acc);
  __syncthreads();
  for (int i = tid; i < 128 * 32; i += 256) {       // stage sender half (K 128..255)
    int r = i >> 5, c4 = (i & 31) * 4;
    float4 v = *(const float4*)(nodesB + (size_t)sendIdx[ebase + r] * DH + c4);
    h16* dst = A + r * 128 + c4;
    dst[0] = (h16)v.x; dst[1] = (h16)v.y; dst[2] = (h16)v.z; dst[3] = (h16)v.w;
  }
  __syncthreads();
  gemm_lds<4, 8>(A, 128, Wt1 + 128, 256, rowA, lane, acc);
  epilog_lds<8>(acc, b1, true, Mid, 128, wave * 16, lane);
  zero_acc8(acc);
  gemm_lds<4, 8>(Mid, 128, Wt2, 128, rowA, lane, acc);
  epilog_lds<8>(acc, b2, true, A, 128, wave * 16, lane);
  zero_acc8(acc);
  gemm_lds<4, 8>(A, 128, Wt3, 128, rowA, lane, acc);
  const int col = lane & 15;
  const int rOff = (lane & 16) ? 8 : 0;
  float b3v[8];
#pragma unroll
  for (int nt = 0; nt < 8; ++nt) b3v[nt] = b3[nt * 16 + col];
#pragma unroll
  for (int r = 0; r < 8; ++r) {
    int e = ebase + wave * 16 + rOff + r;
    float* dst = inb + (size_t)recvIdx[e] * DH + col;
#pragma unroll
    for (int nt = 0; nt < 8; ++nt)
      atomicAdd(dst + nt * 16, acc[nt][r] + b3v[nt]);
  }
}

// --------------------------------------------------------- node update MLP
__global__ void __launch_bounds__(256)
k_node(const float* __restrict__ nodes, const float* __restrict__ inbox,
       const h16* __restrict__ Wt1, const float* __restrict__ b1,
       const h16* __restrict__ Wt2, const float* __restrict__ b2,
       const h16* __restrict__ Wt3, const float* __restrict__ b3,
       float* __restrict__ outNodes)
{
  __shared__ __align__(16) h16 A[128 * 128];
  __shared__ __align__(16) h16 Mid[128 * 128];
  const int b = blockIdx.y;
  const int mbase = blockIdx.x * 128;
  const int tid = threadIdx.x, lane = tid & 31, wave = tid >> 5;
  const float* nodesB = nodes + (size_t)b * MNODES * DH;
  const float* inb = inbox + (size_t)b * MNODES * DH;

  for (int i = tid; i < 128 * 32; i += 256) {
    int r = i >> 5, c4 = (i & 31) * 4;
    int m = mbase + r;
    float4 v = make_float4(0.f, 0.f, 0.f, 0.f);
    if (m < MNODES) v = *(const float4*)(nodesB + (size_t)m * DH + c4);
    h16* dst = A + r * 128 + c4;
    dst[0] = (h16)v.x; dst[1] = (h16)v.y; dst[2] = (h16)v.z; dst[3] = (h16)v.w;
  }
  __syncthreads();
  const int rowA = wave * 16 + (lane & 15);
  v8f acc[8]; zero_acc8(acc);
  gemm_lds<4, 8>(A, 128, Wt1, 256, rowA, lane, acc);
  __syncthreads();
  for (int i = tid; i < 128 * 32; i += 256) {
    int r = i >> 5, c4 = (i & 31) * 4;
    int m = mbase + r;
    float4 v = make_float4(0.f, 0.f, 0.f, 0.f);
    if (m < MNODES) v = *(const float4*)(inb + (size_t)m * DH + c4);
    h16* dst = A + r * 128 + c4;
    dst[0] = (h16)v.x; dst[1] = (h16)v.y; dst[2] = (h16)v.z; dst[3] = (h16)v.w;
  }
  __syncthreads();
  gemm_lds<4, 8>(A, 128, Wt1 + 128, 256, rowA, lane, acc);
  epilog_lds<8>(acc, b1, true, Mid, 128, wave * 16, lane);
  zero_acc8(acc);
  gemm_lds<4, 8>(Mid, 128, Wt2, 128, rowA, lane, acc);
  epilog_lds<8>(acc, b2, true, A, 128, wave * 16, lane);
  zero_acc8(acc);
  gemm_lds<4, 8>(A, 128, Wt3, 128, rowA, lane, acc);
  const int col = lane & 15;
  const int rOff = (lane & 16) ? 8 : 0;
  float* outB = outNodes + (size_t)b * MNODES * DH;
#pragma unroll
  for (int nt = 0; nt < 8; ++nt) {
    float bv = b3[nt * 16 + col];
#pragma unroll
    for (int r = 0; r < 8; ++r) {
      int m = mbase + wave * 16 + rOff + r;
      if (m < MNODES) {
        size_t idx = (size_t)m * DH + nt * 16 + col;
        outB[idx] = nodesB[idx] + acc[nt][r] + bv;       // residual
      }
    }
  }
}

// ----------------------------------------- transpose latents -> f16 [b][d][Mpad]
__global__ void k_tlat(const float* __restrict__ latents, h16* __restrict__ latT)
{
  int i = blockIdx.x * 256 + threadIdx.x;
  if (i >= NBATCH * DH * MPAD) return;
  int mp = i % MPAD; int rest = i / MPAD;
  int d = rest & (DH - 1); int b = rest >> 7;
  float v = (mp < MNODES) ? latents[((size_t)b * MNODES + mp) * DH + d] : 0.f;
  latT[i] = (h16)v;
}

// ----------------------------- z = q_scores @ latents (A built on the fly, K=Mpad)
__global__ void __launch_bounds__(256)
k_z(const float* __restrict__ qpts, const float* __restrict__ stats,
    const float* __restrict__ node_pos, const h16* __restrict__ latT,
    float* __restrict__ zout)
{
  __shared__ float4 stg[128];     // (nx, ny, |n|^2, -) for a chunk of latent nodes
  const int b = blockIdx.y, ntb = blockIdx.x;
  const int tid = threadIdx.x, lane = tid & 31, wave = tid >> 5;
  const int nRow = ntb * 128 + wave * 16 + (lane & 15);
  const int gi0 = b * NPTS + nRow;
  const float qx = qpts[gi0 * 2], qy = qpts[gi0 * 2 + 1];
  const float mx = stats[gi0 * 2], rs = stats[gi0 * 2 + 1];
  const int kb0 = (lane & 16) ? 8 : 0;
  const int kbB = (lane & 16) ? 8 : 0;
  const int col = lane & 15;
  v8f acc[8]; zero_acc8(acc);
  for (int mb = 0; mb < MPAD / 128; ++mb) {
    __syncthreads();
    if (tid < 128) {
      int m = mb * 128 + tid;
      float nx = 0.f, ny = 0.f, ns = 1e30f;   // pad rows -> w = exp(-huge) = 0
      if (m < MNODES) { nx = node_pos[m * 2]; ny = node_pos[m * 2 + 1]; ns = nx * nx + ny * ny; }
      stg[tid] = make_float4(nx, ny, ns, 0.f);
    }
    __syncthreads();
#pragma unroll
    for (int kc = 0; kc < 4; ++kc) {
      V16U a;
#pragma unroll
      for (int j = 0; j < 8; ++j) {
        float4 p = stg[kc * 32 + kb0 + j];
        a.e[j] = (h16)(__expf(2.f * (qx * p.x + qy * p.y) - p.z - mx) * rs);
        float4 p2 = stg[kc * 32 + 16 + kb0 + j];
        a.e[8 + j] = (h16)(__expf(2.f * (qx * p2.x + qy * p2.y) - p2.z - mx) * rs);
      }
#pragma unroll
      for (int nt = 0; nt < 8; ++nt) {
        V16U bf;
        const h16* pb = latT + (size_t)(b * DH + nt * 16 + col) * MPAD + mb * 128 + kc * 32 + kbB;
        bf.h[0] = *(const v8h*)pb;
        bf.h[1] = *(const v8h*)(pb + 16);
        acc[nt] = __builtin_amdgcn_wmma_f32_16x16x32_f16(false, a.v, false, bf.v,
                                                         (short)0, acc[nt], false, false);
      }
    }
  }
  const int rOff = (lane & 16) ? 8 : 0;
#pragma unroll
  for (int nt = 0; nt < 8; ++nt)
#pragma unroll
    for (int r = 0; r < 8; ++r) {
      int n = ntb * 128 + wave * 16 + rOff + r;
      zout[((size_t)b * NPTS + n) * DH + nt * 16 + col] = acc[nt][r];
    }
}

// ------------------------------------------------------------------- decoder
__global__ void __launch_bounds__(256)
k_decoder(const float* __restrict__ z, const float* __restrict__ qpts,
          const float* __restrict__ W1raw, const float* __restrict__ b1,
          const h16* __restrict__ Wt1, const h16* __restrict__ Wt2,
          const float* __restrict__ b2,
          const h16* __restrict__ Wt3, const float* __restrict__ b3,
          float* __restrict__ out)
{
  __shared__ __align__(16) h16 A[128 * 128];
  __shared__ __align__(16) h16 Mid[128 * 128];
  const int b = blockIdx.y;
  const int base = blockIdx.x * 128;
  const int tid = threadIdx.x, lane = tid & 31, wave = tid >> 5;
  for (int i = tid; i < 128 * 32; i += 256) {
    int r = i >> 5, c4 = (i & 31) * 4;
    int gi = b * NPTS + base + r;
    float4 v = *(const float4*)(z + (size_t)gi * DH + c4);
    h16* dst = A + r * 128 + c4;
    dst[0] = (h16)v.x; dst[1] = (h16)v.y; dst[2] = (h16)v.z; dst[3] = (h16)v.w;
  }
  __syncthreads();
  const int rowA = wave * 16 + (lane & 15);
  const int col = lane & 15;
  const int rOff = (lane & 16) ? 8 : 0;
  v8f acc[8]; zero_acc8(acc);
  gemm_lds<4, 8>(A, 128, Wt1, 128, rowA, lane, acc);     // z part of K (0..127)
#pragma unroll
  for (int nt = 0; nt < 8; ++nt) {                       // analytic q contribution
    int d = nt * 16 + col;
    float w128 = W1raw[128 * 128 + d];
    float w129 = W1raw[129 * 128 + d];
    float bv = b1[d];
#pragma unroll
    for (int r = 0; r < 8; ++r) {
      int gi = b * NPTS + base + wave * 16 + rOff + r;
      float v = acc[nt][r] + bv + qpts[gi * 2] * w128 + qpts[gi * 2 + 1] * w129;
      Mid[(wave * 16 + rOff + r) * 128 + d] = (h16)fmaxf(v, 0.f);
    }
  }
  zero_acc8(acc);
  gemm_lds<4, 8>(Mid, 128, Wt2, 128, rowA, lane, acc);
  epilog_lds<8>(acc, b2, true, A, 128, wave * 16, lane);
  v8f acc3[1]; { v8f zz = {}; acc3[0] = zz; }
  gemm_lds<4, 1>(A, 128, Wt3, 128, rowA, lane, acc3);    // N padded 2 -> 16
  if (col < 2) {
    float bv = b3[col];
#pragma unroll
    for (int r = 0; r < 8; ++r) {
      int n = base + wave * 16 + rOff + r;
      out[((size_t)b * NPTS + n) * 2 + col] = acc3[0][r] + bv;
    }
  }
}

// =============================================================== host launcher
extern "C" void kernel_launch(void* const* d_in, const int* in_sizes, int n_in,
                              void* d_out, int out_size, void* d_ws, size_t ws_size,
                              hipStream_t stream)
{
  (void)n_in; (void)out_size; (void)ws_size;
  // Resolve param flattening order: insertion (encoder first) vs alphabetical (blocks first)
  int encB, decB, blkB;
  if (in_sizes[0] == 2 * 128) { encB = 0; decB = 6; blkB = 12; }
  else                        { blkB = 0; decB = 96; encB = 102; }
  const float *encW[3], *encBias[3], *decW[3], *decBias[3];
  for (int l = 0; l < 3; ++l) {
    encW[l]    = (const float*)d_in[encB + 2 * l];
    encBias[l] = (const float*)d_in[encB + 2 * l + 1];
    decW[l]    = (const float*)d_in[decB + 2 * l];
    decBias[l] = (const float*)d_in[decB + 2 * l + 1];
  }
  const float *msgW[8][3], *msgB[8][3], *nodW[8][3], *nodB[8][3];
  for (int s = 0; s < 8; ++s)
    for (int l = 0; l < 3; ++l) {
      msgW[s][l] = (const float*)d_in[blkB + s * 12 + 2 * l];
      msgB[s][l] = (const float*)d_in[blkB + s * 12 + 2 * l + 1];
      nodW[s][l] = (const float*)d_in[blkB + s * 12 + 6 + 2 * l];
      nodB[s][l] = (const float*)d_in[blkB + s * 12 + 6 + 2 * l + 1];
    }
  const float* x        = (const float*)d_in[108];
  const float* sIn      = (const float*)d_in[109];
  const float* q        = (const float*)d_in[110];
  const float* node_pos = (const float*)d_in[111];
  const int*   senders  = (const int*)d_in[112];
  const int*   receivers= (const int*)d_in[113];

  // workspace carve-up
  size_t off = 0;
  char* base = (char*)d_ws;
  auto take = [&](size_t bytes) -> void* {
    void* r = base + off;
    off += (bytes + 255) & ~(size_t)255;
    return r;
  };
  h16* encWt2 = (h16*)take(128 * 128 * sizeof(h16));
  h16* encWt3 = (h16*)take(128 * 128 * sizeof(h16));
  h16* decWt1 = (h16*)take(128 * 128 * sizeof(h16));
  h16* decWt2 = (h16*)take(128 * 128 * sizeof(h16));
  h16* decWt3 = (h16*)take(16 * 128 * sizeof(h16));
  h16 *mWt1[8], *mWt2[8], *mWt3[8], *nWt1[8], *nWt2[8], *nWt3[8];
  for (int s = 0; s < 8; ++s) {
    mWt1[s] = (h16*)take(128 * 256 * sizeof(h16));
    mWt2[s] = (h16*)take(128 * 128 * sizeof(h16));
    mWt3[s] = (h16*)take(128 * 128 * sizeof(h16));
    nWt1[s] = (h16*)take(128 * 256 * sizeof(h16));
    nWt2[s] = (h16*)take(128 * 128 * sizeof(h16));
    nWt3[s] = (h16*)take(128 * 128 * sizeof(h16));
  }
  h16*   embT   = (h16*)take((size_t)NBATCH * DH * NPTS * sizeof(h16));
  float* statsX = (float*)take((size_t)NBATCH * NPTS * 2 * sizeof(float));
  float* statsQ = (float*)take((size_t)NBATCH * NPTS * 2 * sizeof(float));
  float* nodesA = (float*)take((size_t)NBATCH * MNODES * DH * sizeof(float));
  float* nodesB = (float*)take((size_t)NBATCH * MNODES * DH * sizeof(float));
  float* inbox  = (float*)take((size_t)NBATCH * MNODES * DH * sizeof(float));
  h16*   latT   = (h16*)take((size_t)NBATCH * DH * MPAD * sizeof(h16));
  float* zbuf   = (float*)take((size_t)NBATCH * NPTS * DH * sizeof(float));

  auto prep = [&](const float* W, int K, int N, int Kpad, int Npad, h16* out) {
    int total = Npad * Kpad;
    k_prep_wt<<<(total + 255) / 256, 256, 0, stream>>>(W, K, N, Kpad, Npad, out);
  };
  prep(encW[1], 128, 128, 128, 128, encWt2);
  prep(encW[2], 128, 128, 128, 128, encWt3);
  prep(decW[0], 128, 128, 128, 128, decWt1);   // only z rows; q rows handled in-kernel
  prep(decW[1], 128, 128, 128, 128, decWt2);
  prep(decW[2], 128, 2, 128, 16, decWt3);
  for (int s = 0; s < 8; ++s) {
    prep(msgW[s][0], 256, 128, 256, 128, mWt1[s]);
    prep(msgW[s][1], 128, 128, 128, 128, mWt2[s]);
    prep(msgW[s][2], 128, 128, 128, 128, mWt3[s]);
    prep(nodW[s][0], 256, 128, 256, 128, nWt1[s]);
    prep(nodW[s][1], 128, 128, 128, 128, nWt2[s]);
    prep(nodW[s][2], 128, 128, 128, 128, nWt3[s]);
  }

  k_rowstats<<<NBATCH * NPTS / 8, 256, 0, stream>>>(x, node_pos, statsX);
  k_encoder<<<NBATCH * NPTS / 128, 256, 0, stream>>>(
      sIn, encW[0], encBias[0], encWt2, encBias[1], encWt3, encBias[2], embT);
  k_latents<<<dim3((MNODES + 127) / 128, NBATCH), 256, 0, stream>>>(
      x, statsX, node_pos, embT, nodesA);

  float* cur = nodesA;
  float* nxt = nodesB;
  for (int s = 0; s < 8; ++s) {
    hipMemsetAsync(inbox, 0, (size_t)NBATCH * MNODES * DH * sizeof(float), stream);
    k_edge<<<dim3(NEDGE / 128, NBATCH), 256, 0, stream>>>(
        cur, receivers, senders,
        mWt1[s], msgB[s][0], mWt2[s], msgB[s][1], mWt3[s], msgB[s][2], inbox);
    k_node<<<dim3((MNODES + 127) / 128, NBATCH), 256, 0, stream>>>(
        cur, inbox,
        nWt1[s], nodB[s][0], nWt2[s], nodB[s][1], nWt3[s], nodB[s][2], nxt);
    float* t = cur; cur = nxt; nxt = t;
  }

  k_rowstats<<<NBATCH * NPTS / 8, 256, 0, stream>>>(q, node_pos, statsQ);
  k_tlat<<<(NBATCH * DH * MPAD + 255) / 256, 256, 0, stream>>>(cur, latT);
  k_z<<<dim3(NPTS / 128, NBATCH), 256, 0, stream>>>(q, statsQ, node_pos, latT, zbuf);
  k_decoder<<<dim3(NPTS / 128, NBATCH), 256, 0, stream>>>(
      zbuf, q, decW[0], decBias[0], decWt1, decWt2, decBias[1], decWt3, decBias[2],
      (float*)d_out);
}
